// InductiveTemporalScorer_61486751809648
// MI455X (gfx1250) — matrix operand order repaired and checked by reading, not compiled, not run on hardware
//
#include <hip/hip_runtime.h>
#include <hip/hip_bf16.h>

// ---------------- problem constants (from reference) ----------------
#define NN   20000
#define TTT  64
#define FF   128
#define BBQ  8192
#define KK   32
#define QQ   (3*BBQ)
#define WWIN 64
#define DDIM 16
#define MDIM 64
#define HDIM 128
#define HHH  64
#define QPB  8          // queries per workgroup in aggregation kernel

typedef __attribute__((ext_vector_type(16))) _Float16 v16h;
typedef __attribute__((ext_vector_type(8)))  _Float16 v8h;
typedef __attribute__((ext_vector_type(8)))  float    v8f;
typedef __attribute__((ext_vector_type(4)))  int      v4i;

#define AS1 __attribute__((address_space(1)))
#define AS3 __attribute__((address_space(3)))

#ifndef __has_builtin
#define __has_builtin(x) 0
#endif
#if __has_builtin(__builtin_amdgcn_global_load_async_to_lds_b128)
#define HAVE_ASYNC_LDS 1
#else
#define HAVE_ASYNC_LDS 0
#endif

#if HAVE_ASYNC_LDS
__device__ __forceinline__ void async_cp16(const void* g, void* l) {
  __builtin_amdgcn_global_load_async_to_lds_b128((AS1 v4i*)g, (AS3 v4i*)l, 0, 0);
}
__device__ __forceinline__ void async_wait0() {
#if __has_builtin(__builtin_amdgcn_s_wait_asynccnt)
  __builtin_amdgcn_s_wait_asynccnt(0);
#else
  asm volatile("s_wait_asynccnt 0x0" ::: "memory");
#endif
}
#endif

__device__ __forceinline__ v8f vzero8f() {
  v8f z = {0.f,0.f,0.f,0.f,0.f,0.f,0.f,0.f};
  return z;
}
__device__ __forceinline__ v8h vzero8h() {
  v8h z = {(_Float16)0,(_Float16)0,(_Float16)0,(_Float16)0,
           (_Float16)0,(_Float16)0,(_Float16)0,(_Float16)0};
  return z;
}

__device__ __forceinline__ v8f wmma16(v16h a, v16h b, v8f c) {
  return __builtin_amdgcn_wmma_f32_16x16x32_f16(false, a, false, b, (short)0, c, false, false);
}

// B fragment from PRE-PACKED buffer: frag (ks,nt) stored as 32 lanes x 16 halves contiguous.
__device__ __forceinline__ v16h bfragP(const _Float16* P, int nTiles, int ks, int nt, int lane) {
  const _Float16* p = P + (((size_t)ks * nTiles + nt) * 32 + lane) * 16;
  v8h lo = *(const v8h*)p;
  v8h hi = *(const v8h*)(p + 8);
  v16h b;
#pragma unroll
  for (int i = 0; i < 8; ++i) { b[i] = lo[i]; b[8 + i] = hi[i]; }
  return b;
}

// A fragment (16x32 f16) from row-major f16 LDS tile lds[rows][ldk], tile at (m0, k0).
// lanes 0-15: row=m0+lane, K=k0+{0..7,16..23}; lanes 16-31: row=m0+lane-16, K=k0+{8..15,24..31}
__device__ __forceinline__ v16h afrag(const _Float16* lds, int ldk, int m0, int k0, int lane) {
  int half = lane >> 4;
  int row  = m0 + (lane & 15);
  const _Float16* p = lds + row * ldk + k0 + half * 8;
  v8h lo = *(const v8h*)(p);
  v8h hi = *(const v8h*)(p + 16);
  v16h a;
#pragma unroll
  for (int i = 0; i < 8; ++i) { a[i] = lo[i]; a[i + 8] = hi[i]; }
  return a;
}

// ---------------- prep kernels ----------------
// Pack fp32 weight W[Krows][ldn] into WMMA-B-layout f16 fragments.
// blockIdx.x = fragId = ks*nTiles + nt ; 32 threads (one per lane).
// Fragment layout per lane: lanes 0-15: col=nt*16+lane, K=ks*32+{0..7,16..23};
//                           lanes 16-31: col=nt*16+lane-16, K=ks*32+{8..15,24..31}
__global__ void k_packB(const float* __restrict__ W, int ldn, int nTiles,
                        _Float16* __restrict__ out) {
  int frag = blockIdx.x;
  int ks = frag / nTiles, nt = frag % nTiles;
  int lane = threadIdx.x;
  int half = lane >> 4;
  int col  = nt * 16 + (lane & 15);
  int kb   = ks * 32 + half * 8;
  _Float16* d = out + ((size_t)frag * 32 + lane) * 16;
#pragma unroll
  for (int i = 0; i < 16; ++i) d[i] = (_Float16)W[(kb + i + (i >= 8 ? 8 : 0)) * ldn + col];
}

// D[d][o] = delta_emb[d] @ W1[128:144, o] + b1[o]   (65 x 64, fp32)
__global__ void k_dtab(const float* __restrict__ demb, const float* __restrict__ w1,
                       const float* __restrict__ b1, float* __restrict__ dtab) {
  int d = blockIdx.x, o = threadIdx.x;
  float s = b1[o];
#pragma unroll
  for (int j = 0; j < DDIM; ++j) s += demb[d * DDIM + j] * w1[(FF + j) * MDIM + o];
  dtab[d * MDIM + o] = s;
}

// gather x_vt = x_tensor[node_ids[q], times[q]] -> f16 [Q,128]
__global__ __launch_bounds__(128) void k_xvt(const float* __restrict__ x,
                                             const int* __restrict__ nid,
                                             const int* __restrict__ tm,
                                             _Float16* __restrict__ xvt) {
  int q = blockIdx.x, t = threadIdx.x;
  size_t src = ((size_t)nid[q] * TTT + tm[q]) * FF + t;
  xvt[(size_t)q * FF + t] = (_Float16)x[src];
}

// ---------------- aggregation kernel (per-neighbor msg MLP + masked mean) ----------------
__global__ __launch_bounds__(128) void k_agg(
    const float* __restrict__ x, const int* __restrict__ times,
    const int* __restrict__ u_in, const int* __restrict__ tau_in, const unsigned char* __restrict__ mk_in,
    const int* __restrict__ u_out, const int* __restrict__ tau_out, const unsigned char* __restrict__ mk_out,
    const _Float16* __restrict__ p_w1x_in, const _Float16* __restrict__ p_w2_in,
    const float* __restrict__ dtab_in, const float* __restrict__ b2_in,
    const _Float16* __restrict__ p_w1x_out, const _Float16* __restrict__ p_w2_out,
    const float* __restrict__ dtab_out, const float* __restrict__ b2_out,
    _Float16* __restrict__ Min, _Float16* __restrict__ Mout)
{
  __shared__ _Float16 As[32 * 136];   // gathered rows, f16, padded (row stride 272B)
  __shared__ _Float16 Hs[32 * 72];    // hidden after layer1+relu
  __shared__ int   deltas[KK];
  __shared__ float maskf[KK];

  const int tid = threadIdx.x, lane = tid & 31, w = tid >> 5;
  const int q0 = blockIdx.x * QPB;

  for (int dir = 0; dir < 2; ++dir) {
    const int*           U   = dir ? u_out   : u_in;
    const int*           TAU = dir ? tau_out : tau_in;
    const unsigned char* MK  = dir ? mk_out  : mk_in;
    const _Float16*      PW1 = dir ? p_w1x_out : p_w1x_in;
    const _Float16*      PW2 = dir ? p_w2_out  : p_w2_in;
    const float*         DT  = dir ? dtab_out : dtab_in;
    const float*         B2  = dir ? b2_out  : b2_in;
    _Float16*            MD  = dir ? Mout    : Min;
    const int sgn = dir ? -1 : 1;

    // weight fragments held in registers for all QPB queries (coalesced 32B/lane loads)
    v16h bf1[4], bf2[2];
#pragma unroll
    for (int ks = 0; ks < 4; ++ks) bf1[ks] = bfragP(PW1, 4, ks, w, lane);
#pragma unroll
    for (int ks = 0; ks < 2; ++ks) bf2[ks] = bfragP(PW2, 4, ks, w, lane);

    for (int qi = 0; qi < QPB; ++qi) {
      const int q  = q0 + qi;
      const int tq = times[q];

      if (tid < KK) {
        int tau = TAU[q * KK + tid];
        int d = sgn * (tq - tau);
        d = d < 0 ? 0 : (d > WWIN ? WWIN : d);
        deltas[tid] = d;
        maskf[tid] = MK[q * KK + tid] ? 1.0f : 0.0f;
      }
      // gather: thread -> (row = tid/4, 32-float chunk = tid%4); skip masked rows
      {
        const int r  = tid >> 2;
        const int c4 = (tid & 3) * 32;
        const int uu  = U[q * KK + r];
        const int tau = TAU[q * KK + r];
        const bool mk = MK[q * KK + r] != 0;
        _Float16* dst = As + r * 136 + c4;
        if (mk) {
          const float4* s4 = (const float4*)(x + ((size_t)uu * TTT + tau) * FF + c4);
#pragma unroll
          for (int i = 0; i < 4; ++i) {
            float4 a = s4[2 * i], b = s4[2 * i + 1];
            v8h h = {(_Float16)a.x, (_Float16)a.y, (_Float16)a.z, (_Float16)a.w,
                     (_Float16)b.x, (_Float16)b.y, (_Float16)b.z, (_Float16)b.w};
            *(v8h*)(dst + i * 8) = h;
          }
        } else {
          v8h z = vzero8h();
#pragma unroll
          for (int i = 0; i < 4; ++i) *(v8h*)(dst + i * 8) = z;
        }
        // prefetch next query's gather row (random 512B rows -> hide HBM latency)
        if (qi + 1 < QPB) {
          const int qn = q + 1;
          const int un  = U[qn * KK + r];
          const int tn  = TAU[qn * KK + r];
          if (MK[qn * KK + r]) {
            const float* pf = x + ((size_t)un * TTT + tn) * FF + c4;
            __builtin_prefetch(pf, 0, 1);
          }
        }
      }
      __syncthreads();

      // GEMM1: [32,128] x [128,64]; preload all A-frags, then back-to-back WMMAs
      v8f acc0 = vzero8f(), acc1 = vzero8f();
      {
        v16h a0[4], a1[4];
#pragma unroll
        for (int ks = 0; ks < 4; ++ks) {
          a0[ks] = afrag(As, 136, 0,  ks * 32, lane);
          a1[ks] = afrag(As, 136, 16, ks * 32, lane);
        }
#pragma unroll
        for (int ks = 0; ks < 4; ++ks) {
          acc0 = wmma16(a0[ks], bf1[ks], acc0);
          acc1 = wmma16(a1[ks], bf1[ks], acc1);
        }
      }
      const int coln  = w * 16 + (lane & 15);
      const int rbase = (lane >> 4) * 8;
#pragma unroll
      for (int r = 0; r < 8; ++r) {
        int row = rbase + r;
        float v = acc0[r] + DT[deltas[row] * MDIM + coln];
        Hs[row * 72 + coln] = (_Float16)(v > 0.f ? v : 0.f);
        row = 16 + rbase + r;
        v = acc1[r] + DT[deltas[row] * MDIM + coln];
        Hs[row * 72 + coln] = (_Float16)(v > 0.f ? v : 0.f);
      }
      __syncthreads();

      // GEMM2: [32,64] x [64,64]
      acc0 = vzero8f(); acc1 = vzero8f();
      {
        v16h a0[2], a1[2];
#pragma unroll
        for (int ks = 0; ks < 2; ++ks) {
          a0[ks] = afrag(Hs, 72, 0,  ks * 32, lane);
          a1[ks] = afrag(Hs, 72, 16, ks * 32, lane);
        }
#pragma unroll
        for (int ks = 0; ks < 2; ++ks) {
          acc0 = wmma16(a0[ks], bf2[ks], acc0);
          acc1 = wmma16(a1[ks], bf2[ks], acc1);
        }
      }
      // masked mean over 32 neighbors
      float cnt = 0.f;
#pragma unroll
      for (int k = 0; k < KK; ++k) cnt += maskf[k];
      float p = 0.f;
#pragma unroll
      for (int r = 0; r < 8; ++r)
        p += maskf[rbase + r] * acc0[r] + maskf[16 + rbase + r] * acc1[r];
      p += __shfl_xor(p, 16);
      if (lane < 16) {
        float inv = cnt > 0.f ? 1.f / cnt : 0.f;
        float mv  = p * inv + (cnt > 0.f ? B2[coln] : 0.f);
        MD[(size_t)q * MDIM + coln] = (_Float16)mv;
      }
      __syncthreads();
    }
  }
}

// ---------------- combine kernel: [Q,256] -> relu -> [Q,128] -> relu -> h[Q,64] ----------------
__global__ __launch_bounds__(128) void k_combine(
    const _Float16* __restrict__ xvt, const _Float16* __restrict__ min_,
    const _Float16* __restrict__ mout_,
    const _Float16* __restrict__ p_cw1, const float* __restrict__ cb1,
    const _Float16* __restrict__ p_cw2, const float* __restrict__ cb2,
    _Float16* __restrict__ hbuf)
{
  __shared__ _Float16 Ac[64 * 264];
  __shared__ _Float16 Hc[64 * 136];
  const int tid = threadIdx.x, lane = tid & 31, w = tid >> 5;
  const int q0 = blockIdx.x * 64;

  {   // build A = [xvt | m_in | m_out], 64 rows x 256 halves
    const int row = tid >> 1, ch = tid & 1;
    const int q = q0 + row;
    _Float16* dst = Ac + row * 264 + ch * 128;
#if HAVE_ASYNC_LDS
    if (ch == 0) {
      const _Float16* g = xvt + (size_t)q * FF;
#pragma unroll
      for (int i = 0; i < 16; ++i) async_cp16(g + i * 8, dst + i * 8);
    } else {
      const _Float16* g1 = min_  + (size_t)q * MDIM;
      const _Float16* g2 = mout_ + (size_t)q * MDIM;
#pragma unroll
      for (int i = 0; i < 8; ++i) async_cp16(g1 + i * 8, dst + i * 8);
#pragma unroll
      for (int i = 0; i < 8; ++i) async_cp16(g2 + i * 8, dst + 64 + i * 8);
    }
#else
    v8h* d = (v8h*)dst;
    if (ch == 0) {
      const v8h* s = (const v8h*)(xvt + (size_t)q * FF);
#pragma unroll
      for (int i = 0; i < 16; ++i) d[i] = s[i];
    } else {
      const v8h* s1 = (const v8h*)(min_  + (size_t)q * MDIM);
      const v8h* s2 = (const v8h*)(mout_ + (size_t)q * MDIM);
#pragma unroll
      for (int i = 0; i < 8; ++i) d[i] = s1[i];
#pragma unroll
      for (int i = 0; i < 8; ++i) d[8 + i] = s2[i];
    }
#endif
  }
#if HAVE_ASYNC_LDS
  async_wait0();
#endif
  __syncthreads();

  const int rbase = (lane >> 4) * 8;
  // GEMM1: K=256 (8 ksteps), N=128 -> wave handles 2 n-tiles
  for (int nt = 2 * w; nt <= 2 * w + 1; ++nt) {
    v8f acc[4];
#pragma unroll
    for (int mt = 0; mt < 4; ++mt) acc[mt] = vzero8f();
    for (int ks = 0; ks < 8; ++ks) {
      v16h bf = bfragP(p_cw1, 8, ks, nt, lane);
#pragma unroll
      for (int mt = 0; mt < 4; ++mt) {
        v16h a = afrag(Ac, 264, mt * 16, ks * 32, lane);
        acc[mt] = wmma16(a, bf, acc[mt]);
      }
    }
    const int coln = nt * 16 + (lane & 15);
    const float bias = cb1[coln];
#pragma unroll
    for (int mt = 0; mt < 4; ++mt)
#pragma unroll
      for (int r = 0; r < 8; ++r) {
        float v = acc[mt][r] + bias;
        Hc[(mt * 16 + rbase + r) * 136 + coln] = (_Float16)(v > 0.f ? v : 0.f);
      }
  }
  __syncthreads();

  // GEMM2: K=128 (4 ksteps), N=64 -> wave handles 1 n-tile
  {
    v8f acc[4];
#pragma unroll
    for (int mt = 0; mt < 4; ++mt) acc[mt] = vzero8f();
    for (int ks = 0; ks < 4; ++ks) {
      v16h bf = bfragP(p_cw2, 4, ks, w, lane);
#pragma unroll
      for (int mt = 0; mt < 4; ++mt) {
        v16h a = afrag(Hc, 136, mt * 16, ks * 32, lane);
        acc[mt] = wmma16(a, bf, acc[mt]);
      }
    }
    const int coln = w * 16 + (lane & 15);
    const float bias = cb2[coln];
#pragma unroll
    for (int mt = 0; mt < 4; ++mt)
#pragma unroll
      for (int r = 0; r < 8; ++r) {
        float v = acc[mt][r] + bias;
        hbuf[(size_t)(q0 + mt * 16 + rbase + r) * HHH + coln] = (_Float16)(v > 0.f ? v : 0.f);
      }
  }
}

// ---------------- scorer kernel: feat[B,193] -> relu(128) -> scalar ----------------
__global__ __launch_bounds__(128) void k_score(
    const _Float16* __restrict__ hbuf, const _Float16* __restrict__ p_sw1,
    const float* __restrict__ sw1f, const float* __restrict__ sb1,
    const float* __restrict__ sw2, const float* __restrict__ sb2,
    const float* __restrict__ d_norm, float* __restrict__ out)
{
  __shared__ _Float16 As[64 * 200];
  __shared__ float    H2[64 * 133];
  const int tid = threadIdx.x, lane = tid & 31, w = tid >> 5;
  const int b0 = blockIdx.x * 64;

  {   // feat row b = 192 contiguous halves of hbuf starting at row 3b (reshape(B,192))
    const int row = tid >> 1, ch = tid & 1;
    const _Float16* g = hbuf + (size_t)(3 * (b0 + row)) * HHH + ch * 96;
    _Float16* dst = As + row * 200 + ch * 96;
#if HAVE_ASYNC_LDS
#pragma unroll
    for (int i = 0; i < 12; ++i) async_cp16(g + i * 8, dst + i * 8);
#else
    const v8h* s = (const v8h*)g;
    v8h* d = (v8h*)dst;
#pragma unroll
    for (int i = 0; i < 12; ++i) d[i] = s[i];
#endif
  }
#if HAVE_ASYNC_LDS
  async_wait0();
#endif
  __syncthreads();

  const int rbase = (lane >> 4) * 8;
  for (int nt = 2 * w; nt <= 2 * w + 1; ++nt) {
    v8f acc[4];
#pragma unroll
    for (int mt = 0; mt < 4; ++mt) acc[mt] = vzero8f();
    for (int ks = 0; ks < 6; ++ks) {          // K = 192
      v16h bf = bfragP(p_sw1, 8, ks, nt, lane);
#pragma unroll
      for (int mt = 0; mt < 4; ++mt) {
        v16h a = afrag(As, 200, mt * 16, ks * 32, lane);
        acc[mt] = wmma16(a, bf, acc[mt]);
      }
    }
    const int coln = nt * 16 + (lane & 15);
    const float w192 = sw1f[192 * HDIM + coln];   // d_norm column of sw1 (fp32)
    const float bias = sb1[coln];
#pragma unroll
    for (int mt = 0; mt < 4; ++mt)
#pragma unroll
      for (int r = 0; r < 8; ++r) {
        const int row = mt * 16 + rbase + r;
        float v = acc[mt][r] + d_norm[b0 + row] * w192 + bias;
        H2[row * 133 + coln] = v > 0.f ? v : 0.f;
      }
  }
  __syncthreads();

  if (tid < 64) {
    float s = sb2[0];
#pragma unroll 8
    for (int o = 0; o < HDIM; ++o) s += H2[tid * 133 + o] * sw2[o];
    out[b0 + tid] = s;
  }
}

// ---------------- launcher ----------------
extern "C" void kernel_launch(void* const* d_in, const int* in_sizes, int n_in,
                              void* d_out, int out_size, void* d_ws, size_t ws_size,
                              hipStream_t stream) {
  const float* x         = (const float*)d_in[0];
  const float* d_norm    = (const float*)d_in[1];
  const float* delta_emb = (const float*)d_in[2];
  const float* piw1 = (const float*)d_in[3];  const float* pib1 = (const float*)d_in[4];
  const float* piw2 = (const float*)d_in[5];  const float* pib2 = (const float*)d_in[6];
  const float* pow1 = (const float*)d_in[7];  const float* pob1 = (const float*)d_in[8];
  const float* pow2 = (const float*)d_in[9];  const float* pob2 = (const float*)d_in[10];
  const float* cw1  = (const float*)d_in[11]; const float* cb1  = (const float*)d_in[12];
  const float* cw2  = (const float*)d_in[13]; const float* cb2  = (const float*)d_in[14];
  const float* sw1  = (const float*)d_in[15]; const float* sb1  = (const float*)d_in[16];
  const float* sw2  = (const float*)d_in[17]; const float* sb2  = (const float*)d_in[18];
  const int* node_ids = (const int*)d_in[19]; const int* times = (const int*)d_in[20];
  const int* in_u   = (const int*)d_in[21];   const int* in_tau  = (const int*)d_in[22];
  const unsigned char* in_mask  = (const unsigned char*)d_in[23];
  const int* out_u  = (const int*)d_in[24];   const int* out_tau = (const int*)d_in[25];
  const unsigned char* out_mask = (const unsigned char*)d_in[26];
  float* out = (float*)d_out;

  // workspace layout (256B aligned slots)
  char* ws = (char*)d_ws;
  size_t off = 0;
  auto alloc = [&](size_t bytes) { void* p = ws + off; off = (off + bytes + 255) & ~(size_t)255; return p; };
  // packed WMMA-B fragments: 32 lanes x 16 halves each
  _Float16* p_w1x_in  = (_Float16*)alloc((size_t)4 * 4 * 512 * 2);   // 4 ksteps x 4 ntiles
  _Float16* p_w2_in   = (_Float16*)alloc((size_t)2 * 4 * 512 * 2);
  _Float16* p_w1x_out = (_Float16*)alloc((size_t)4 * 4 * 512 * 2);
  _Float16* p_w2_out  = (_Float16*)alloc((size_t)2 * 4 * 512 * 2);
  _Float16* p_cw1     = (_Float16*)alloc((size_t)8 * 8 * 512 * 2);   // 8 ksteps x 8 ntiles
  _Float16* p_cw2     = (_Float16*)alloc((size_t)4 * 4 * 512 * 2);
  _Float16* p_sw1     = (_Float16*)alloc((size_t)6 * 8 * 512 * 2);   // 6 ksteps x 8 ntiles
  float*    dtab_in   = (float*)alloc((size_t)(WWIN + 1) * MDIM * 4);
  float*    dtab_out  = (float*)alloc((size_t)(WWIN + 1) * MDIM * 4);
  _Float16* xvt   = (_Float16*)alloc((size_t)QQ * FF * 2);
  _Float16* m_in  = (_Float16*)alloc((size_t)QQ * MDIM * 2);
  _Float16* m_out = (_Float16*)alloc((size_t)QQ * MDIM * 2);
  _Float16* hbuf  = (_Float16*)alloc((size_t)QQ * HHH * 2);
  (void)ws_size; (void)in_sizes; (void)n_in; (void)out_size;

  // pack weights into WMMA-B fragment layout (fp32 -> f16)
  k_packB<<<4 * 4, 32, 0, stream>>>(piw1, MDIM, 4, p_w1x_in);   // rows 0..127 of [144,64]
  k_packB<<<2 * 4, 32, 0, stream>>>(piw2, MDIM, 4, p_w2_in);
  k_packB<<<4 * 4, 32, 0, stream>>>(pow1, MDIM, 4, p_w1x_out);
  k_packB<<<2 * 4, 32, 0, stream>>>(pow2, MDIM, 4, p_w2_out);
  k_packB<<<8 * 8, 32, 0, stream>>>(cw1, HDIM, 8, p_cw1);
  k_packB<<<4 * 4, 32, 0, stream>>>(cw2, HHH, 4, p_cw2);
  k_packB<<<6 * 8, 32, 0, stream>>>(sw1, HDIM, 8, p_sw1);       // rows 0..191 of [193,128]

  k_dtab<<<WWIN + 1, MDIM, 0, stream>>>(delta_emb, piw1, pib1, dtab_in);
  k_dtab<<<WWIN + 1, MDIM, 0, stream>>>(delta_emb, pow1, pob1, dtab_out);

  k_xvt<<<QQ, 128, 0, stream>>>(x, node_ids, times, xvt);

  k_agg<<<QQ / QPB, 128, 0, stream>>>(
      x, times,
      in_u, in_tau, in_mask, out_u, out_tau, out_mask,
      p_w1x_in, p_w2_in, dtab_in, pib2,
      p_w1x_out, p_w2_out, dtab_out, pob2,
      m_in, m_out);

  k_combine<<<QQ / 64, 128, 0, stream>>>(xvt, m_in, m_out, p_cw1, cb1, p_cw2, cb2, hbuf);

  k_score<<<BBQ / 64, 128, 0, stream>>>(hbuf, p_sw1, sw1, sb1, sw2, sb2, d_norm, out);
}
